// AutoReg_48619029791066
// MI455X (gfx1250) — matrix operand order repaired and checked by reading, not compiled, not run on hardware
//
#include <hip/hip_runtime.h>
#include <stdint.h>

#define D_SZ   112
#define H3     768
#define INGRU  537
#define CBM_SZ 536
#define LOG2PI 1.8378770664093453f

typedef __attribute__((ext_vector_type(16))) __bf16 v16bf;
typedef __attribute__((ext_vector_type(8)))  float  v8f;

union Frag16 { uint32_t u[8]; v16bf v; };

// Global tile numbering inside the packed-bf16 weight region (tile = 32 lanes * 8 dwords)
#define TWHH 0     // 8 kk * 48 nn = 384 tiles  (W_hh^T, 256x768)
#define TW1  384   // 8 kk * 16 nn = 128 tiles  (w1_h,  256x256)
#define TW2  512   // 128 tiles                 (w2,    256x256)
#define TW3  640   // 8 kk * 4 nn  = 32 tiles   (w3,    256x64 zero-padded)
#define NTILES 672

__device__ __forceinline__ uint16_t f2bf(float f) {
  uint32_t x = __float_as_uint(f);
  uint32_t r = x + 0x7FFFu + ((x >> 16) & 1u);   // round-to-nearest-even
  return (uint16_t)(r >> 16);
}

__device__ __forceinline__ v8f wmma_bf16(v16bf a, v16bf b, v8f c) {
  return __builtin_amdgcn_wmma_f32_16x16x32_bf16(false, a, false, b, (short)0, c, false, false);
}

// Load WMMA B fragment: 8 contiguous dwords per lane from prepacked weights (2x b128)
__device__ __forceinline__ void load_bfrag(const uint32_t* packed, int tileIdx, int lane, Frag16& bf) {
  const uint4* p = (const uint4*)(packed + ((size_t)tileIdx * 32 + lane) * 8);
  uint4 x0 = p[0], x1 = p[1];
  bf.u[0] = x0.x; bf.u[1] = x0.y; bf.u[2] = x0.z; bf.u[3] = x0.w;
  bf.u[4] = x1.x; bf.u[5] = x1.y; bf.u[6] = x1.z; bf.u[7] = x1.w;
}

// Load 8 A fragments (16x256 bf16 row-major in LDS), contiguous k order (k = 16*half + s)
__device__ __forceinline__ void load_afrags(const uint16_t* buf, int lane, Frag16 af[8]) {
  int r = lane & 15, hh = lane >> 4;
  const uint32_t* row = (const uint32_t*)(buf + r * 256);
#pragma unroll
  for (int kk = 0; kk < 8; ++kk) {
#pragma unroll
    for (int v = 0; v < 8; ++v) {
      int K = kk * 32 + hh * 16 + v * 2;
      af[kk].u[v] = row[K >> 1];
    }
  }
}

// ---------------- Kernel 1: prepack weights to bf16 WMMA-B fragment order --------------
__global__ void prepack_kernel(const float* __restrict__ whh, const float* __restrict__ w1,
                               const float* __restrict__ w2, const float* __restrict__ w3,
                               uint32_t* __restrict__ packed) {
  int gid = blockIdx.x * blockDim.x + threadIdx.x;
  if (gid >= NTILES * 32) return;
  int tile = gid >> 5, lane = gid & 31;
  int nlo = lane & 15, khalf = lane >> 4;
  uint32_t out[8];
  if (tile < TW1) {                       // W_hh^T : B[K][n] = whh[n*256 + K]
    int kk = tile / 48, nn = tile % 48;
    int n = nn * 16 + nlo;
#pragma unroll
    for (int v = 0; v < 8; ++v) {
      int K = kk * 32 + khalf * 16 + 2 * v;
      uint32_t lo = f2bf(whh[(size_t)n * 256 + K]);
      uint32_t hi = f2bf(whh[(size_t)n * 256 + K + 1]);
      out[v] = lo | (hi << 16);
    }
  } else if (tile < TW2) {                // w1_h : B[K][n] = w1[K*256 + n]
    int t2 = tile - TW1;
    int kk = t2 / 16, nn = t2 % 16;
    int n = nn * 16 + nlo;
#pragma unroll
    for (int v = 0; v < 8; ++v) {
      int K = kk * 32 + khalf * 16 + 2 * v;
      uint32_t lo = f2bf(w1[(size_t)K * 256 + n]);
      uint32_t hi = f2bf(w1[(size_t)(K + 1) * 256 + n]);
      out[v] = lo | (hi << 16);
    }
  } else if (tile < TW3) {                // w2
    int t2 = tile - TW2;
    int kk = t2 / 16, nn = t2 % 16;
    int n = nn * 16 + nlo;
#pragma unroll
    for (int v = 0; v < 8; ++v) {
      int K = kk * 32 + khalf * 16 + 2 * v;
      uint32_t lo = f2bf(w2[(size_t)K * 256 + n]);
      uint32_t hi = f2bf(w2[(size_t)(K + 1) * 256 + n]);
      out[v] = lo | (hi << 16);
    }
  } else {                                // w3 (256x60, pad n>=60 with 0)
    int t2 = tile - TW3;
    int kk = t2 / 4, nn = t2 % 4;
    int n = nn * 16 + nlo;
#pragma unroll
    for (int v = 0; v < 8; ++v) {
      int K = kk * 32 + khalf * 16 + 2 * v;
      uint32_t lo = (n < 60) ? f2bf(w3[(size_t)K * 60 + n]) : 0u;
      uint32_t hi = (n < 60) ? f2bf(w3[(size_t)(K + 1) * 60 + n]) : 0u;
      out[v] = lo | (hi << 16);
    }
  }
  uint32_t* dst = packed + ((size_t)tile * 32 + lane) * 8;
#pragma unroll
  for (int v = 0; v < 8; ++v) dst[v] = out[v];
}

// ---------------- Kernel 2: step-invariant constants (one-off f32 GEMMs) ---------------
__global__ void const_gemm_kernel(const float* __restrict__ cc, const float* __restrict__ bb,
                                  const float* __restrict__ mm, const float* __restrict__ gwih,
                                  const float* __restrict__ gbih, const float* __restrict__ w1,
                                  const float* __restrict__ b1,
                                  float* __restrict__ gi_const, float* __restrict__ mlp_const) {
  __shared__ float s_cbm[CBM_SZ];
  int b = blockIdx.x, tid = threadIdx.x;
  for (int k = tid; k < CBM_SZ; k += 256) {
    float v;
    if (k < 312)      v = cc[(size_t)b * 312 + k];
    else if (k < 424) v = bb[(size_t)b * 112 + (k - 312)];
    else              v = mm[(size_t)b * 112 + (k - 424)];
    s_cbm[k] = v;
  }
  __syncthreads();
  for (int j = tid; j < H3; j += 256) {
    float acc = gbih[j];
    const float* wr = gwih + (size_t)j * INGRU + 1;
    for (int k = 0; k < CBM_SZ; ++k) acc += s_cbm[k] * wr[k];
    gi_const[(size_t)b * H3 + j] = acc;
  }
  {
    int j = tid;  // exactly 256 columns
    float acc = b1[j];
    for (int k = 0; k < CBM_SZ; ++k) acc += s_cbm[k] * w1[(size_t)(256 + k) * 256 + j];
    mlp_const[(size_t)b * 256 + j] = acc;
  }
}

// ---------------- Kernel 3: 112-step GRU+MLP scan, WMMA bf16 ---------------------------
// 128 WGs x 256 threads; each WG owns a 16-row batch tile for the whole scan.
// All activations live in LDS; A fragments are reloaded from LDS per GEMM so no large
// register arrays survive across barriers. asm memory clobbers stop LICM from hoisting
// the (t-invariant) packed-weight loads into spilled registers.
__global__ void __launch_bounds__(256)
scan_kernel(const float* __restrict__ z, const uint32_t* __restrict__ packed,
            const float* __restrict__ gi_const, const float* __restrict__ mlp_const,
            const float* __restrict__ gwih, const float* __restrict__ gbhh,
            const float* __restrict__ b2, const float* __restrict__ b3,
            float* __restrict__ loglike) {
  extern __shared__ char smem[];
  float*    s_gh  = (float*)smem;                 // 16*768
  float*    s_h   = s_gh + 16 * H3;               // 16*256
  float*    s_wz  = s_h + 16 * 256;               // 768
  float*    s_zt  = s_wz + H3;                    // 16
  float*    s_p   = s_zt + 16;                    // 16*64
  uint16_t* s_hbf = (uint16_t*)(s_p + 16 * 64);   // 16*256 bf16
  uint16_t* s_a1  = s_hbf + 16 * 256;
  uint16_t* s_a2  = s_a1 + 16 * 256;

  int tid = threadIdx.x;
  int lane = tid & 31, wave = tid >> 5;
  int nlo = lane & 15, mbase = (lane >> 4) * 8;
  int b0 = blockIdx.x * 16;

  for (int idx = tid; idx < H3; idx += 256) s_wz[idx] = gwih[(size_t)idx * INGRU];  // W_ih[:,0]
  for (int idx = tid; idx < 16 * 256; idx += 256) {
    s_h[idx] = 0.f;            // h0 = 0
    s_hbf[idx] = 0;            // bf16(0)
  }
  __syncthreads();

  for (int t = 0; t < D_SZ; ++t) {
    // ---- GEMM1: gh = h @ W_hh^T + b_hh  (48 n-tiles, 6 per wave) ----
    {
      Frag16 af[8];
      load_afrags(s_hbf, lane, af);            // h_t
#pragma unroll 1
      for (int i = 0; i < 6; ++i) {
        asm volatile("" ::: "memory");         // keep weight loads inside the loop
        int nn = wave * 6 + i;
        int col = nn * 16 + nlo;
        float bias = gbhh[col];
        v8f c;
#pragma unroll
        for (int j = 0; j < 8; ++j) c[j] = bias;
#pragma unroll
        for (int kk = 0; kk < 8; ++kk) {
          Frag16 bf;
          load_bfrag(packed, TWHH + kk * 48 + nn, lane, bf);
          c = wmma_bf16(af[kk].v, bf.v, c);
        }
#pragma unroll
        for (int j = 0; j < 8; ++j) s_gh[(mbase + j) * H3 + col] = c[j];
      }
    }
    if (tid < 16) s_zt[tid] = (t == 0) ? -1.f : z[(size_t)(b0 + tid) * D_SZ + t - 1];
    __syncthreads();

    // ---- GRU gates (elementwise, thread = column) ----
    {
      int j = tid;
      float wz0 = s_wz[j], wz1 = s_wz[256 + j], wz2 = s_wz[512 + j];
      for (int r = 0; r < 16; ++r) {
        float ztr = s_zt[r];
        const float* gi = gi_const + (size_t)(b0 + r) * H3;
        float ir = gi[j]        + ztr * wz0;
        float iz = gi[256 + j]  + ztr * wz1;
        float in = gi[512 + j]  + ztr * wz2;
        float hr = s_gh[r * H3 + j];
        float hz = s_gh[r * H3 + 256 + j];
        float hn = s_gh[r * H3 + 512 + j];
        float rg = 1.f / (1.f + __expf(-(ir + hr)));
        float ug = 1.f / (1.f + __expf(-(iz + hz)));
        float ng = tanhf(in + rg * hn);
        float hv = (1.f - ug) * ng + ug * s_h[r * 256 + j];
        s_h[r * 256 + j] = hv;                 // h_{t+1} (f32 state)
        s_hbf[r * 256 + j] = f2bf(hv);         // h_{t+1} (bf16 for WMMA A)
      }
    }
    __syncthreads();

    // ---- GEMM2: a1 = tanh(h_new @ w1_h + mlp_const)  (16 n-tiles, 2/wave) ----
    {
      Frag16 af[8];
      load_afrags(s_hbf, lane, af);            // h_{t+1}
#pragma unroll 1
      for (int i = 0; i < 2; ++i) {
        asm volatile("" ::: "memory");
        int nn = wave * 2 + i;
        int col = nn * 16 + nlo;
        v8f c;
#pragma unroll
        for (int j = 0; j < 8; ++j) c[j] = mlp_const[(size_t)(b0 + mbase + j) * 256 + col];
#pragma unroll
        for (int kk = 0; kk < 8; ++kk) {
          Frag16 bf;
          load_bfrag(packed, TW1 + kk * 16 + nn, lane, bf);
          c = wmma_bf16(af[kk].v, bf.v, c);
        }
#pragma unroll
        for (int j = 0; j < 8; ++j) s_a1[(mbase + j) * 256 + col] = f2bf(tanhf(c[j]));
      }
    }
    __syncthreads();

    // ---- GEMM3: a2 = tanh(a1 @ w2 + b2) ----
    {
      Frag16 af[8];
      load_afrags(s_a1, lane, af);
#pragma unroll 1
      for (int i = 0; i < 2; ++i) {
        asm volatile("" ::: "memory");
        int nn = wave * 2 + i;
        int col = nn * 16 + nlo;
        v8f c;
#pragma unroll
        for (int j = 0; j < 8; ++j) c[j] = b2[col];
#pragma unroll
        for (int kk = 0; kk < 8; ++kk) {
          Frag16 bf;
          load_bfrag(packed, TW2 + kk * 16 + nn, lane, bf);
          c = wmma_bf16(af[kk].v, bf.v, c);
        }
#pragma unroll
        for (int j = 0; j < 8; ++j) s_a2[(mbase + j) * 256 + col] = f2bf(tanhf(c[j]));
      }
    }
    __syncthreads();

    // ---- GEMM4: p = a2 @ w3 + b3   (4 n-tiles, waves 0..3) ----
    {
      Frag16 af[8];
      load_afrags(s_a2, lane, af);
      if (wave < 4) {
        asm volatile("" ::: "memory");
        int nn = wave;
        int col = nn * 16 + nlo;
        v8f c;
#pragma unroll
        for (int j = 0; j < 8; ++j) c[j] = (col < 60) ? b3[col] : 0.f;
#pragma unroll
        for (int kk = 0; kk < 8; ++kk) {
          Frag16 bf;
          load_bfrag(packed, TW3 + kk * 4 + nn, lane, bf);
          c = wmma_bf16(af[kk].v, bf.v, c);
        }
#pragma unroll
        for (int j = 0; j < 8; ++j) s_p[(mbase + j) * 64 + col] = c[j];
      }
    }
    __syncthreads();

    // ---- MDN log-likelihood per row ----
    if (tid < 16) {
      int r = tid;
      float target = z[(size_t)(b0 + r) * D_SZ + t];
      const float* pr = s_p + r * 64;
      float mA = -1e30f, mL = -1e30f;
      for (int k = 0; k < 20; ++k) {
        float lg = pr[k], mu = pr[20 + k], ls = pr[40 + k];
        float u = (target - mu) * __expf(-ls);
        float tv = lg - ls - 0.5f * LOG2PI - 0.5f * u * u;
        mA = fmaxf(mA, tv);
        mL = fmaxf(mL, lg);
      }
      float sA = 0.f, sL = 0.f;
      for (int k = 0; k < 20; ++k) {
        float lg = pr[k], mu = pr[20 + k], ls = pr[40 + k];
        float u = (target - mu) * __expf(-ls);
        float tv = lg - ls - 0.5f * LOG2PI - 0.5f * u * u;
        sA += __expf(tv - mA);
        sL += __expf(lg - mL);
      }
      loglike[(size_t)(b0 + r) * D_SZ + t] = (mA + __logf(sA)) - (mL + __logf(sL));
    }
    __syncthreads();
  }
}

// ---------------- Kernel 4: descending-sort mask + masked sum --------------------------
__global__ void finalize_kernel(const float* __restrict__ bb, const float* __restrict__ mm,
                                const float* __restrict__ loglike, float* __restrict__ out) {
  __shared__ float s_q[D_SZ];
  __shared__ float s_part[D_SZ];
  int row = blockIdx.x, tid = threadIdx.x;
  if (tid < D_SZ) s_q[tid] = mm[(size_t)row * D_SZ + tid] * (1.f - bb[(size_t)row * D_SZ + tid]);
  __syncthreads();
  if (tid < D_SZ) {
    float qj = s_q[tid];
    int rank = 0;
    for (int k = 0; k < D_SZ; ++k) {
      float qk = s_q[k];
      rank += (qk > qj) || (qk == qj && k < tid);
    }
    s_part[tid] = qj * loglike[(size_t)row * D_SZ + rank];
  }
  __syncthreads();
  if (tid == 0) {
    float acc = 0.f;
    for (int k = 0; k < D_SZ; ++k) acc += s_part[k];
    out[row] = acc;
  }
}

extern "C" void kernel_launch(void* const* d_in, const int* in_sizes, int n_in,
                              void* d_out, int out_size, void* d_ws, size_t ws_size,
                              hipStream_t stream) {
  (void)in_sizes; (void)n_in; (void)out_size; (void)ws_size;
  const float* z    = (const float*)d_in[0];
  const float* cc   = (const float*)d_in[1];
  const float* bb   = (const float*)d_in[2];
  const float* mm   = (const float*)d_in[3];
  const float* gwih = (const float*)d_in[4];
  const float* whh  = (const float*)d_in[5];
  const float* gbih = (const float*)d_in[6];
  const float* gbhh = (const float*)d_in[7];
  const float* w1   = (const float*)d_in[8];
  const float* b1   = (const float*)d_in[9];
  const float* w2   = (const float*)d_in[10];
  const float* b2   = (const float*)d_in[11];
  const float* w3   = (const float*)d_in[12];
  const float* b3   = (const float*)d_in[13];
  float* out = (float*)d_out;

  float* ws        = (float*)d_ws;
  float* gi_const  = ws;                                   // 2048*768 f32
  float* mlp_const = gi_const + (size_t)2048 * 768;        // 2048*256 f32
  float* loglike   = mlp_const + (size_t)2048 * 256;       // 2048*112 f32
  uint32_t* packed = (uint32_t*)(loglike + (size_t)2048 * 112);  // 672 tiles * 256 dwords

  prepack_kernel<<<(NTILES * 32 + 255) / 256, 256, 0, stream>>>(whh, w1, w2, w3, packed);
  const_gemm_kernel<<<2048, 256, 0, stream>>>(cc, bb, mm, gwih, gbih, w1, b1, gi_const, mlp_const);

  size_t smem = (size_t)(16 * H3 + 16 * 256 + H3 + 16 + 16 * 64) * sizeof(float)
              + (size_t)3 * 16 * 256 * sizeof(uint16_t);   // ~97 KB (fits 320 KB/WGP LDS)
  scan_kernel<<<128, 256, smem, stream>>>(z, packed, gi_const, mlp_const, gwih, gbhh, b2, b3, loglike);

  finalize_kernel<<<2048, 128, 0, stream>>>(bb, mm, loglike, out);
}
